// FullAttention_8048768713212
// MI455X (gfx1250) — matrix operand order repaired and checked by reading, not compiled, not run on hardware
//
#include <hip/hip_runtime.h>
#include <hip/hip_bf16.h>

// ---------------------------------------------------------------------------
// Fused MHA forward for MI455X (gfx1250, wave32, WMMA, bf16 matmul / f32 acc).
//   B=4, S=2048, D=1024, H=16, DH=64
//   Stage 0: convert q/k/v -> bf16, W -> transposed bf16
//   Stage 1: projections via 32x64-tile WMMA GEMM (all b128 fragment loads)
//   Stage 2: flash attention, 16 queries/wave, key blocks of 32
// ---------------------------------------------------------------------------

#define B_ 4
#define S_ 2048
#define D_ 1024
#define H_ 16
#define DH_ 64

typedef __attribute__((ext_vector_type(16))) __bf16 v16bf;
typedef __attribute__((ext_vector_type(8)))  float  v8f;

union AFrag { int i[8]; v16bf v; };   // 16x32 A or 32x16 B bf16 fragment
union CFrag { float f[8]; v8f v; };   // 16x16 f32 C/D fragment

__device__ __forceinline__ unsigned short f2bf(float f) {
    unsigned u = __float_as_uint(f);
    return (unsigned short)((u + 0x7FFFu + ((u >> 16) & 1u)) >> 16);   // RNE
}
__device__ __forceinline__ int pack2(unsigned short lo, unsigned short hi) {
    return (int)((unsigned)lo | ((unsigned)hi << 16));
}

// A-fragment (16x32 bf16): per lane two contiguous 16B chunks at +8h, +16+8h elems
__device__ __forceinline__ void ld_a(AFrag& f, const unsigned short* p) {
    const int4 x = *(const int4*)(p);
    const int4 y = *(const int4*)(p + 16);
    f.i[0] = x.x; f.i[1] = x.y; f.i[2] = x.z; f.i[3] = x.w;
    f.i[4] = y.x; f.i[5] = y.y; f.i[6] = y.z; f.i[7] = y.w;
}
// B-fragment (32x16 bf16): per lane two contiguous 16B chunks at +8h, +16+8h elems
__device__ __forceinline__ void ld_b(AFrag& f, const unsigned short* p) {
    const int4 x = *(const int4*)(p);
    const int4 y = *(const int4*)(p + 16);
    f.i[0] = x.x; f.i[1] = x.y; f.i[2] = x.z; f.i[3] = x.w;
    f.i[4] = y.x; f.i[5] = y.y; f.i[6] = y.z; f.i[7] = y.w;
}

__device__ __forceinline__ v8f wmma_bf16(const v16bf& a, const v16bf& b, const v8f& c) {
    return __builtin_amdgcn_wmma_f32_16x16x32_bf16(false, a, false, b, (short)0, c,
                                                   false, false);
}

// ---------------------------------------------------------------------------
// Elementwise f32 -> packed bf16 (pairs).  n2 = number of f32 pairs.
// ---------------------------------------------------------------------------
__global__ void __launch_bounds__(256)
cvt_bf16_kernel(const float* __restrict__ in, unsigned int* __restrict__ out, int n2) {
    const int idx = blockIdx.x * 256 + threadIdx.x;
    if (idx < n2) {
        const float2 v = ((const float2*)in)[idx];
        out[idx] = (unsigned int)pack2(f2bf(v.x), f2bf(v.y));
    }
}

// W (DxD f32, row-major) -> Wt (DxD bf16, transposed: Wt[col*D + k] = W[k*D + col])
__global__ void __launch_bounds__(256)
cvt_wt_kernel(const float* __restrict__ in, unsigned short* __restrict__ out) {
    const int idx = blockIdx.x * 256 + threadIdx.x;   // D*D threads
    const int k = idx & (D_ - 1);
    const int c = idx >> 10;
    out[(size_t)c * D_ + k] = f2bf(in[(size_t)k * D_ + c]);
}

// ---------------------------------------------------------------------------
// Projection: Y = X(8192x1024 bf16) @ W(1024x1024, Wt bf16 transposed).
// One wave per 32x64 output tile (2 A-frags share each B-frag -> 8 WMMA/chunk).
// transposed==0 : out (B,H,S,DH) ; transposed==1 : out (B,H,DH,S) (for V)
// ---------------------------------------------------------------------------
__global__ void __launch_bounds__(128)
proj_kernel(const unsigned short* __restrict__ Xb,
            const unsigned short* __restrict__ Wt,
            unsigned short* __restrict__ out, int transposed) {
    const int lane = threadIdx.x & 31;
    const int wave = threadIdx.x >> 5;
    const int flat = blockIdx.x * 4 + wave;          // 0..4095
    const int tileM = flat >> 4;                     // 0..255  (32 rows each)
    const int tileN = flat & 15;                     // 0..15   (64 cols each)
    const int m  = lane & 15;
    const int hh = lane >> 4;

    const unsigned short* xrow0 = Xb + (size_t)(tileM * 32 + m) * D_;
    const unsigned short* xrow1 = xrow0 + (size_t)16 * D_;

    CFrag acc[2][4];
#pragma unroll
    for (int u = 0; u < 2; ++u)
#pragma unroll
        for (int c = 0; c < 4; ++c)
#pragma unroll
            for (int r = 0; r < 8; ++r) acc[u][c].f[r] = 0.0f;

    for (int k0 = 0; k0 < D_; k0 += 32) {
        AFrag a0, a1;
        ld_a(a0, xrow0 + k0 + 8 * hh);
        ld_a(a1, xrow1 + k0 + 8 * hh);
#pragma unroll
        for (int c = 0; c < 4; ++c) {
            const int coln = tileN * 64 + c * 16 + m;
            AFrag bf;
            ld_b(bf, Wt + (size_t)coln * D_ + k0 + 8 * hh);
            acc[0][c].v = wmma_bf16(a0.v, bf.v, acc[0][c].v);
            acc[1][c].v = wmma_bf16(a1.v, bf.v, acc[1][c].v);
        }
    }

#pragma unroll
    for (int u = 0; u < 2; ++u)
#pragma unroll
        for (int c = 0; c < 4; ++c)
#pragma unroll
            for (int r = 0; r < 8; ++r) {
                const int gr  = tileM * 32 + u * 16 + r + 8 * hh;  // row in (B*S)
                const int b   = gr >> 11;
                const int s   = gr & (S_ - 1);
                const int col = tileN * 64 + c * 16 + m;
                const int h   = col >> 6;
                const int dh  = col & 63;
                size_t idx = transposed
                    ? ((size_t)((b * H_ + h) * DH_ + dh)) * S_ + s
                    : ((size_t)((b * H_ + h) * S_  + s)) * DH_ + dh;
                out[idx] = f2bf(acc[u][c].f[r]);
            }
}

// ---------------------------------------------------------------------------
// Flash attention.  Block = 128 threads (4 waves); wave owns 16 queries.
// Grid = B*H*(S/64) = 2048.  Key loop in blocks of 32.
// ---------------------------------------------------------------------------
__global__ void __launch_bounds__(128)
attn_kernel(const unsigned short* __restrict__ Qp,
            const unsigned short* __restrict__ Kp,
            const unsigned short* __restrict__ VpT,
            const float* __restrict__ mask,
            float* __restrict__ out) {
    __shared__ __align__(16) unsigned short pstage[4][16][32];

    const int lane = threadIdx.x & 31;
    const int wave = threadIdx.x >> 5;
    const int qblk = blockIdx.x & 31;              // S/64 = 32
    const int bh   = blockIdx.x >> 5;              // 0..63
    const int h    = bh & 15;
    const int b    = bh >> 4;
    const int q0   = qblk * 64 + wave * 16;
    const int m    = lane & 15;
    const int hh   = lane >> 4;

    const unsigned short* qbase = Qp  + (size_t)(b * H_ + h) * S_ * DH_;
    const unsigned short* kbase = Kp  + (size_t)(b * H_ + h) * S_ * DH_;
    const unsigned short* vbase = VpT + (size_t)(b * H_ + h) * DH_ * S_;

    AFrag qa[2];
    {
        const unsigned short* qrow = qbase + (size_t)(q0 + m) * DH_;
        ld_a(qa[0], qrow + 8 * hh);
        ld_a(qa[1], qrow + 32 + 8 * hh);
    }

    float mrow[8], lrow[8];
#pragma unroll
    for (int r = 0; r < 8; ++r) { mrow[r] = -INFINITY; lrow[r] = 0.0f; }
    CFrag o[4];
#pragma unroll
    for (int c = 0; c < 4; ++c)
#pragma unroll
        for (int r = 0; r < 8; ++r) o[c].f[r] = 0.0f;

    const float scale = 0.125f;   // 1/sqrt(DH)

    for (int kb = 0; kb < S_; kb += 32) {
        if (kb + 32 < S_) {       // global_prefetch_b8 for next K/V tiles
            __builtin_prefetch(kbase + (size_t)(kb + 32 + m) * DH_, 0, 1);
            __builtin_prefetch(vbase + (size_t)m * S_ + kb + 32, 0, 1);
        }

        // ---- scores: two 16x16 tiles ----
        CFrag s[2];
#pragma unroll
        for (int t = 0; t < 2; ++t) {
            const unsigned short* krow =
                kbase + (size_t)(kb + 16 * t + m) * DH_ + 8 * hh;
            AFrag kb0, kb1;
            ld_b(kb0, krow);        // d in [0,32)
            ld_b(kb1, krow + 32);   // d in [32,64)
#pragma unroll
            for (int r = 0; r < 8; ++r) s[t].f[r] = 0.0f;
            s[t].v = wmma_bf16(qa[0].v, kb0.v, s[t].v);
            s[t].v = wmma_bf16(qa[1].v, kb1.v, s[t].v);
#pragma unroll
            for (int r = 0; r < 8; ++r) {
                const float mv =
                    mask[(size_t)(q0 + r + 8 * hh) * S_ + kb + 16 * t + m];
                s[t].f[r] *= scale * mv;   // multiplicative mask before softmax
            }
        }

        // ---- online softmax (row = r + 8*hh, cols across 16-lane half) ----
        float alpha[8];
#pragma unroll
        for (int r = 0; r < 8; ++r) {
            float mx = fmaxf(s[0].f[r], s[1].f[r]);
#pragma unroll
            for (int d = 1; d < 16; d <<= 1) mx = fmaxf(mx, __shfl_xor(mx, d));
            const float mnew = fmaxf(mrow[r], mx);
            alpha[r] = __expf(mrow[r] - mnew);
            mrow[r]  = mnew;
            s[0].f[r] = __expf(s[0].f[r] - mnew);
            s[1].f[r] = __expf(s[1].f[r] - mnew);
            float sum = s[0].f[r] + s[1].f[r];
#pragma unroll
            for (int d = 1; d < 16; d <<= 1) sum += __shfl_xor(sum, d);
            lrow[r] = lrow[r] * alpha[r] + sum;
        }
#pragma unroll
        for (int c = 0; c < 4; ++c)
#pragma unroll
            for (int r = 0; r < 8; ++r) o[c].f[r] *= alpha[r];

        // ---- transpose P (C layout -> A layout) through LDS ----
#pragma unroll
        for (int t = 0; t < 2; ++t)
#pragma unroll
            for (int r = 0; r < 8; ++r)
                pstage[wave][r + 8 * hh][m + 16 * t] = f2bf(s[t].f[r]);
        __syncthreads();

        AFrag pa;
        ld_a(pa, &pstage[wave][m][0] + 8 * hh);

        // ---- O += P @ V ----
#pragma unroll
        for (int c = 0; c < 4; ++c) {
            AFrag vb;
            ld_b(vb, vbase + (size_t)(16 * c + m) * S_ + kb + 8 * hh);
            o[c].v = wmma_bf16(pa.v, vb.v, o[c].v);
        }
        __syncthreads();
    }

    // ---- epilogue: divide by l, write (B,S,D) f32 ----
#pragma unroll
    for (int c = 0; c < 4; ++c)
#pragma unroll
        for (int r = 0; r < 8; ++r) {
            const float val = o[c].f[r] / lrow[r];
            const size_t idx =
                (size_t)(b * S_ + q0 + r + 8 * hh) * D_ + h * 64 + c * 16 + m;
            out[idx] = val;
        }
}

// ---------------------------------------------------------------------------
extern "C" void kernel_launch(void* const* d_in, const int* in_sizes, int n_in,
                              void* d_out, int out_size, void* d_ws, size_t ws_size,
                              hipStream_t stream) {
    (void)in_sizes; (void)n_in; (void)out_size; (void)ws_size;
    const float* q    = (const float*)d_in[0];
    const float* k    = (const float*)d_in[1];
    const float* v    = (const float*)d_in[2];
    const float* mask = (const float*)d_in[3];
    const float* Wq   = (const float*)d_in[4];
    const float* Wk   = (const float*)d_in[5];
    const float* Wv   = (const float*)d_in[6];
    float* out = (float*)d_out;

    const size_t X_ELEMS    = (size_t)B_ * S_ * D_;        // 8388608
    const size_t W_ELEMS    = (size_t)D_ * D_;             // 1048576
    const size_t PROJ_ELEMS = (size_t)B_ * H_ * S_ * DH_;  // 8388608

    unsigned short* ws = (unsigned short*)d_ws;
    unsigned short* Xq  = ws;                    // bf16 copies of inputs
    unsigned short* Xk  = Xq  + X_ELEMS;
    unsigned short* Xv  = Xk  + X_ELEMS;
    unsigned short* Wtq = Xv  + X_ELEMS;         // transposed bf16 weights
    unsigned short* Wtk = Wtq + W_ELEMS;
    unsigned short* Wtv = Wtk + W_ELEMS;
    unsigned short* Qp  = Wtv + W_ELEMS;         // projected heads
    unsigned short* Kp  = Qp  + PROJ_ELEMS;
    unsigned short* VpT = Kp  + PROJ_ELEMS;

    // Stage 0: conversions
    const int n2 = (int)(X_ELEMS / 2);                       // 4194304 pairs
    cvt_bf16_kernel<<<n2 / 256, 256, 0, stream>>>(q, (unsigned int*)Xq, n2);
    cvt_bf16_kernel<<<n2 / 256, 256, 0, stream>>>(k, (unsigned int*)Xk, n2);
    cvt_bf16_kernel<<<n2 / 256, 256, 0, stream>>>(v, (unsigned int*)Xv, n2);
    cvt_wt_kernel<<<(int)(W_ELEMS / 256), 256, 0, stream>>>(Wq, Wtq);
    cvt_wt_kernel<<<(int)(W_ELEMS / 256), 256, 0, stream>>>(Wk, Wtk);
    cvt_wt_kernel<<<(int)(W_ELEMS / 256), 256, 0, stream>>>(Wv, Wtv);

    // Stage 1: projections (4096 wave-tiles of 32x64 -> 1024 blocks)
    proj_kernel<<<1024, 128, 0, stream>>>(Xq, Wtq, Qp,  0);
    proj_kernel<<<1024, 128, 0, stream>>>(Xk, Wtk, Kp,  0);
    proj_kernel<<<1024, 128, 0, stream>>>(Xv, Wtv, VpT, 1);

    // Stage 2: attention (B*H*(S/64) = 2048 blocks, 4 waves each)
    attn_kernel<<<2048, 128, 0, stream>>>(Qp, Kp, VpT, mask, out);
}
